// GNNEncoder_60181081751573
// MI455X (gfx1250) — matrix operand order repaired and checked by reading, not compiled, not run on hardware
//
#include <hip/hip_runtime.h>
#include <stdint.h>

// ---------------------------------------------------------------------------
// Types for CDNA5 WMMA (wave32): V_WMMA_F32_16X16X32_BF16
// ---------------------------------------------------------------------------
typedef __bf16 bf16_t;
typedef bf16_t bf16x8  __attribute__((ext_vector_type(8)));
typedef bf16_t bf16x16 __attribute__((ext_vector_type(16)));
typedef float  f32x8   __attribute__((ext_vector_type(8)));

__device__ __forceinline__ unsigned short f2bf(float x) {
    union { float f; unsigned int u; } v; v.f = x;
    unsigned int r = v.u + 0x7FFFu + ((v.u >> 16) & 1u);   // round-to-nearest-even
    return (unsigned short)(r >> 16);
}

// ---------------------------------------------------------------------------
// Degree: one atomic per edge (computed once; identical for both layers)
// ---------------------------------------------------------------------------
__global__ void deg_kernel(const int* __restrict__ dst, int E, float* __restrict__ deg) {
    int t = blockIdx.x * blockDim.x + threadIdx.x;
    if (t < E) atomicAdd(&deg[dst[t]], 1.0f);
}

// ---------------------------------------------------------------------------
// Scatter-add of feature rows over edges. One wave == one edge:
// 32 lanes x float4 = 512B coalesced read of the source row, atomics to dst.
// ---------------------------------------------------------------------------
__global__ void scatter_kernel(const float* __restrict__ xsrc, const int* __restrict__ ei,
                               int E, float* __restrict__ agg) {
    long t = (long)blockIdx.x * blockDim.x + threadIdx.x;
    if (t >= (long)E * 32) return;
    int e = (int)(t >> 5);
    int c = ((int)t & 31) << 2;
    int s = ei[e];
    int d = ei[E + e];
    const float4 v = *(const float4*)(xsrc + (long)s * 128 + c);
    float* o = agg + (long)d * 128 + c;
    atomicAdd(o + 0, v.x);
    atomicAdd(o + 1, v.y);
    atomicAdd(o + 2, v.z);
    atomicAdd(o + 3, v.w);
}

// ---------------------------------------------------------------------------
// mean (divide by max(deg,1)) + convert to bf16. Pad rows have deg==0,agg==0.
// ---------------------------------------------------------------------------
__global__ void mean_bf16_kernel(const float* __restrict__ agg, const float* __restrict__ deg,
                                 unsigned short* __restrict__ out, long nelem) {
    long t = (long)blockIdx.x * blockDim.x + threadIdx.x;
    if (t >= nelem) return;
    float inv = 1.0f / fmaxf(deg[t >> 7], 1.0f);
    out[t] = f2bf(agg[t] * inv);
}

// f32 -> bf16 with zero fill for padded rows
__global__ void cvt_bf16_kernel(const float* __restrict__ in, unsigned short* __restrict__ out,
                                long nreal, long ntotal) {
    long t = (long)blockIdx.x * blockDim.x + threadIdx.x;
    if (t >= ntotal) return;
    out[t] = (t < nreal) ? f2bf(in[t]) : (unsigned short)0;
}

// ---------------------------------------------------------------------------
// Build stacked bf16 weights for one (layer, dst-type):
//   seg s < nrel : Wl[rel_s]            (128x128, row n = out-channel)
//   seg nrel     : sum_s Wr[rel_s]      (x_dst term collapsed across relations)
//   bias         : sum_s b[rel_s]
// ---------------------------------------------------------------------------
__global__ void prep_w_kernel(const float* __restrict__ Wl, const float* __restrict__ bv,
                              const float* __restrict__ Wr,
                              int r0, int r1, int r2, int nrel,
                              unsigned short* __restrict__ wstack, float* __restrict__ bstack) {
    int t = blockIdx.x * blockDim.x + threadIdx.x;
    if (t >= 128 * 128) return;
    int rel[3] = { r0, r1, r2 };
    float wr = 0.f;
    for (int s = 0; s < nrel; ++s) {
        wstack[s * 16384 + t] = f2bf(Wl[(long)rel[s] * 16384 + t]);
        wr += Wr[(long)rel[s] * 16384 + t];
    }
    wstack[nrel * 16384 + t] = f2bf(wr);
    if (t < 128) {
        float bb = 0.f;
        for (int s = 0; s < nrel; ++s) bb += bv[rel[s] * 128 + t];
        bstack[t] = bb;
    }
}

// ---------------------------------------------------------------------------
// One 128-deep segment of the stacked GEMM: 4 x WMMA 16x16x32 (bf16 -> f32).
// A (16x32 bf16): lane L<16 -> row L, elems
// kt..kt+15; lanes>=16 -> kt+16..kt+31.
// B (32x16): lane n<16 -> W-row n elems kt..kt+15; lanes>=16 -> kt+16..kt+31.
// Abase/Bbase derive directly from kernel-arg pointers so the compiler keeps
// global address space (global_load_b128, LOADcnt only).
// ---------------------------------------------------------------------------
__device__ __forceinline__ void gemm_seg128(const unsigned short* __restrict__ Abase,
                                            const unsigned short* __restrict__ Bbase,
                                            int lhi, f32x8& acc) {
#pragma unroll
    for (int kt = 0; kt < 128; kt += 32) {
        bf16x8 alo = *(const bf16x8*)(Abase + kt + lhi * 16);
        bf16x8 ahi = *(const bf16x8*)(Abase + kt + lhi * 16 + 8);
        bf16x16 af;
#pragma unroll
        for (int i = 0; i < 8; ++i) { af[i] = alo[i]; af[i + 8] = ahi[i]; }
        bf16x16 bfv = *(const bf16x16*)(Bbase + kt + lhi * 16);
        acc = __builtin_amdgcn_wmma_f32_16x16x32_bf16(
                false, af, false, bfv, (short)0, acc, false, false);
    }
}

// ---------------------------------------------------------------------------
// Fused stacked GEMM + bias + ReLU. NSEG is compile-time: fully unrolled
// 4*NSEG WMMAs, no pointer arrays / runtime segment selects.
// Block: 256 threads = 8 waves; wave w owns N-chunk [16w,16w+16); block owns
// 16 M-rows.
// ---------------------------------------------------------------------------
template <int NSEG>
__global__ __launch_bounds__(256) void wmma_gemm_relu_kernel(
        const unsigned short* __restrict__ A0, const unsigned short* __restrict__ A1,
        const unsigned short* __restrict__ A2, const unsigned short* __restrict__ A3,
        const unsigned short* __restrict__ Wst, const float* __restrict__ bias,
        float* __restrict__ out, int M) {
    const int wave = threadIdx.x >> 5;
    const int lane = threadIdx.x & 31;
    const int lmod = lane & 15;
    const int lhi  = lane >> 4;
    const long mbase = (long)blockIdx.x * 16;

    const long rowOff = (mbase + lmod) * 128;                 // A row for this lane
    const long bOff   = (long)(wave * 16 + lmod) * 128;       // W row for this lane

    f32x8 acc = { 0.f, 0.f, 0.f, 0.f, 0.f, 0.f, 0.f, 0.f };

    gemm_seg128(A0 + rowOff, Wst + 0 * 16384 + bOff, lhi, acc);
    if (NSEG > 1) gemm_seg128(A1 + rowOff, Wst + 1 * 16384 + bOff, lhi, acc);
    if (NSEG > 2) gemm_seg128(A2 + rowOff, Wst + 2 * 16384 + bOff, lhi, acc);
    if (NSEG > 3) gemm_seg128(A3 + rowOff, Wst + 3 * 16384 + bOff, lhi, acc);

    const float bvn = bias[wave * 16 + lmod];
    const int   col = wave * 16 + lmod;
#pragma unroll
    for (int v = 0; v < 8; ++v) {
        long mg = mbase + lhi * 8 + v;        // C/D layout: VGPR v -> M=v (+8 hi lanes)
        if (mg < M) out[mg * 128 + col] = fmaxf(acc[v] + bvn, 0.0f);
    }
}

// ---------------------------------------------------------------------------
// Host-side orchestration
// ---------------------------------------------------------------------------
extern "C" void kernel_launch(void* const* d_in, const int* in_sizes, int n_in,
                              void* d_out, int out_size, void* d_ws, size_t ws_size,
                              hipStream_t stream) {
    (void)n_in; (void)out_size; (void)ws_size;

    const float* xu  = (const float*)d_in[0];
    const float* xb  = (const float*)d_in[1];
    const float* xc  = (const float*)d_in[2];
    const float* W1l = (const float*)d_in[3];
    const float* b1  = (const float*)d_in[4];
    const float* W1r = (const float*)d_in[5];
    const float* W2l = (const float*)d_in[6];
    const float* b2  = (const float*)d_in[7];
    const float* W2r = (const float*)d_in[8];
    // relation order in weights: 0 ub, 1 bu, 2 uu, 3 bb, 4 bc, 5 cb
    const int* EI[6] = { (const int*)d_in[9],  (const int*)d_in[10], (const int*)d_in[11],
                         (const int*)d_in[12], (const int*)d_in[13], (const int*)d_in[14] };
    int EN[6];
    for (int r = 0; r < 6; ++r) EN[r] = in_sizes[9 + r] / 2;

    const int NU = 100000, NB = 50000, NC = 1000;
    const int PU = NU, PB = NB, PC = (NC + 15) & ~15;   // pad rows to 16

    // ---- workspace carve -------------------------------------------------
    char* ws = (char*)d_ws;
    size_t off = 0;
    auto alloc = [&](size_t bytes) -> void* {
        void* p = ws + off;
        off = (off + bytes + 255) & ~(size_t)255;
        return p;
    };

    const size_t degFloats = (size_t)2 * PU + 3 * PB + PC;
    float* degBase = (float*)alloc(degFloats * sizeof(float));
    float* deg_bu = degBase;
    float* deg_uu = deg_bu + PU;
    float* deg_ub = deg_uu + PU;
    float* deg_bb = deg_ub + PB;
    float* deg_cb = deg_bb + PB;
    float* deg_bc = deg_cb + PB;

    const size_t aggRows = (size_t)2 * PU + 3 * PB + PC;
    float* aggBase = (float*)alloc(aggRows * 128 * sizeof(float));
    float* agg_bu = aggBase;
    float* agg_uu = agg_bu + (size_t)PU * 128;
    float* agg_ub = agg_uu + (size_t)PU * 128;
    float* agg_bb = agg_ub + (size_t)PB * 128;
    float* agg_cb = agg_bb + (size_t)PB * 128;
    float* agg_bc = agg_cb + (size_t)PB * 128;

    unsigned short* aghBase = (unsigned short*)alloc(aggRows * 128 * sizeof(unsigned short));
    unsigned short* agh_bu = aghBase;
    unsigned short* agh_uu = agh_bu + (size_t)PU * 128;
    unsigned short* agh_ub = agh_uu + (size_t)PU * 128;
    unsigned short* agh_bb = agh_ub + (size_t)PB * 128;
    unsigned short* agh_cb = agh_bb + (size_t)PB * 128;
    unsigned short* agh_bc = agh_cb + (size_t)PB * 128;

    unsigned short* xbfU = (unsigned short*)alloc((size_t)PU * 128 * 2);
    unsigned short* xbfB = (unsigned short*)alloc((size_t)PB * 128 * 2);
    unsigned short* xbfC = (unsigned short*)alloc((size_t)PC * 128 * 2);

    float* h1u = (float*)alloc((size_t)NU * 128 * 4);
    float* h1b = (float*)alloc((size_t)NB * 128 * 4);
    float* h1c = (float*)alloc((size_t)NC * 128 * 4);

    unsigned short* wst = (unsigned short*)alloc((size_t)2 * 9 * 16384 * 2);
    unsigned short* wst_u1 = wst;
    unsigned short* wst_b1 = wst + (size_t)3 * 16384;
    unsigned short* wst_c1 = wst + (size_t)7 * 16384;
    unsigned short* wst_u2 = wst + (size_t)9 * 16384;
    unsigned short* wst_b2 = wst + (size_t)12 * 16384;
    unsigned short* wst_c2 = wst + (size_t)16 * 16384;
    float* bst = (float*)alloc((size_t)6 * 128 * 4);
    float* bst_u1 = bst, *bst_b1 = bst + 128, *bst_c1 = bst + 256;
    float* bst_u2 = bst + 384, *bst_b2 = bst + 512, *bst_c2 = bst + 640;

    float* out_u = (float*)d_out;
    float* out_b = out_u + (size_t)NU * 128;
    float* out_c = out_b + (size_t)NB * 128;

    auto cdiv = [](long a, long b) -> long { return (a + b - 1) / b; };

    // ---- zero deg + agg (f32) -------------------------------------------
    hipMemsetAsync(degBase, 0, degFloats * sizeof(float), stream);
    hipMemsetAsync(aggBase, 0, aggRows * 128 * sizeof(float), stream);

    // ---- degrees (once) --------------------------------------------------
    float* DEG[6] = { deg_ub, deg_bu, deg_uu, deg_bb, deg_bc, deg_cb };
    for (int r = 0; r < 6; ++r)
        deg_kernel<<<cdiv(EN[r], 256), 256, 0, stream>>>(EI[r] + EN[r], EN[r], DEG[r]);

    // ---- weight prep -----------------------------------------------------
    prep_w_kernel<<<64, 256, 0, stream>>>(W1l, b1, W1r, 1, 2, -1, 2, wst_u1, bst_u1);
    prep_w_kernel<<<64, 256, 0, stream>>>(W1l, b1, W1r, 0, 3,  5, 3, wst_b1, bst_b1);
    prep_w_kernel<<<64, 256, 0, stream>>>(W1l, b1, W1r, 4, -1, -1, 1, wst_c1, bst_c1);
    prep_w_kernel<<<64, 256, 0, stream>>>(W2l, b2, W2r, 1, 2, -1, 2, wst_u2, bst_u2);
    prep_w_kernel<<<64, 256, 0, stream>>>(W2l, b2, W2r, 0, 3,  5, 3, wst_b2, bst_b2);
    prep_w_kernel<<<64, 256, 0, stream>>>(W2l, b2, W2r, 4, -1, -1, 1, wst_c2, bst_c2);

    // per-relation tables (relation order: ub, bu, uu, bb, bc, cb)
    float*          AGG[6] = { agg_ub, agg_bu, agg_uu, agg_bb, agg_bc, agg_cb };
    unsigned short* AGH[6] = { agh_ub, agh_bu, agh_uu, agh_bb, agh_bc, agh_cb };
    long            PD[6]  = { PB, PU, PU, PB, PC, PB };
    const float*    S1[6]  = { xu, xb, xu, xb, xb, xc };      // layer-1 sources
    const float*    S2[6]  = { h1u, h1b, h1u, h1b, h1b, h1c };// layer-2 sources

    for (int layer = 0; layer < 2; ++layer) {
        if (layer == 1)
            hipMemsetAsync(aggBase, 0, aggRows * 128 * sizeof(float), stream);

        // scatter-add + mean/bf16 per relation
        for (int r = 0; r < 6; ++r) {
            const float* src = (layer == 0) ? S1[r] : S2[r];
            scatter_kernel<<<cdiv((long)EN[r] * 32, 256), 256, 0, stream>>>(
                src, EI[r], EN[r], AGG[r]);
        }
        for (int r = 0; r < 6; ++r) {
            long ne = PD[r] * 128;
            mean_bf16_kernel<<<cdiv(ne, 256), 256, 0, stream>>>(AGG[r], DEG[r], AGH[r], ne);
        }

        // x_dst -> bf16 (zero-filled pads)
        const float* su = (layer == 0) ? xu : h1u;
        const float* sb = (layer == 0) ? xb : h1b;
        const float* sc = (layer == 0) ? xc : h1c;
        cvt_bf16_kernel<<<cdiv((long)PU * 128, 256), 256, 0, stream>>>(su, xbfU, (long)NU * 128, (long)PU * 128);
        cvt_bf16_kernel<<<cdiv((long)PB * 128, 256), 256, 0, stream>>>(sb, xbfB, (long)NB * 128, (long)PB * 128);
        cvt_bf16_kernel<<<cdiv((long)PC * 128, 256), 256, 0, stream>>>(sc, xbfC, (long)NC * 128, (long)PC * 128);

        // fused stacked GEMM + bias + ReLU
        unsigned short* wu = (layer == 0) ? wst_u1 : wst_u2;
        unsigned short* wb = (layer == 0) ? wst_b1 : wst_b2;
        unsigned short* wc = (layer == 0) ? wst_c1 : wst_c2;
        float* bu_ = (layer == 0) ? bst_u1 : bst_u2;
        float* bb_ = (layer == 0) ? bst_b1 : bst_b2;
        float* bc_ = (layer == 0) ? bst_c1 : bst_c2;
        float* ou = (layer == 0) ? h1u : out_u;
        float* ob = (layer == 0) ? h1b : out_b;
        float* oc = (layer == 0) ? h1c : out_c;

        wmma_gemm_relu_kernel<3><<<PU / 16, 256, 0, stream>>>(
            agh_bu, agh_uu, xbfU, xbfU, wu, bu_, ou, NU);
        wmma_gemm_relu_kernel<4><<<PB / 16, 256, 0, stream>>>(
            agh_ub, agh_bb, agh_cb, xbfB, wb, bb_, ob, NB);
        wmma_gemm_relu_kernel<2><<<PC / 16, 256, 0, stream>>>(
            agh_bc, xbfC, xbfC, xbfC, wc, bc_, oc, NC);
    }
}
